// Attention0Pooling_87428354277729
// MI455X (gfx1250) — matrix operand order repaired and checked by reading, not compiled, not run on hardware
//
#include <hip/hip_runtime.h>
#include <hip/hip_bf16.h>
#include <math.h>

// ---------------------------------------------------------------------------
// MI455X (gfx1250) cluster-attention pooling block.
// wave32, WMMA f32_16x16x32_f16 everywhere, two-pass softmax over the cluster
// axis (never materializes the 944MB attention matrix). f16 intermediates
// (~70MB) stay L2-resident; register blocking keeps L2 traffic ~1.5 b128
// loads per WMMA in the GEMMs and better in the attention loops.
// ---------------------------------------------------------------------------

typedef __attribute__((ext_vector_type(16))) _Float16 v16h;
typedef __attribute__((ext_vector_type(8)))  _Float16 v8h;
typedef __attribute__((ext_vector_type(8)))  float    v8f;

#define NB     4
#define NTOK   3136      // 56*56
#define NC     1568      // 2*28*28
#define DMODEL 384
#define NH     12
#define DH     32
#define NPJ    784       // NC/2
#define DPROJ  768
#define QSCALE 0.17677669529663687f   // 32^-0.5
#define EPS_LN 1e-6f
#define EPS_TS 1e-6f

__device__ __forceinline__ v16h mk16(v8h lo, v8h hi) {
  v16h r;
#pragma unroll
  for (int i = 0; i < 8; ++i) { r[i] = lo[i]; r[i + 8] = hi[i]; }
  return r;
}

__device__ __forceinline__ v8f zero8() {
  v8f z = {0.f, 0.f, 0.f, 0.f, 0.f, 0.f, 0.f, 0.f};
  return z;
}

__device__ __forceinline__ v8f wmma_f16(v16h a, v16h b, v8f c) {
  return __builtin_amdgcn_wmma_f32_16x16x32_f16(false, a, false, b,
                                                (short)0, c, false, false);
}

// ---------------------------------------------------------------------------
// LayerNorm over 384 (writes f16), one block per row.
// ---------------------------------------------------------------------------
__global__ void ln384_f16(const float* __restrict__ x, const float* __restrict__ g,
                          _Float16* __restrict__ o) {
  __shared__ float sA[128], sB[128];
  int row = blockIdx.x, tid = threadIdx.x;
  const float* xr = x + (size_t)row * DMODEL;
  float v0 = xr[tid], v1 = xr[tid + 128], v2 = xr[tid + 256];
  sA[tid] = v0 + v1 + v2;
  sB[tid] = v0 * v0 + v1 * v1 + v2 * v2;
  __syncthreads();
  for (int st = 64; st > 0; st >>= 1) {
    if (tid < st) { sA[tid] += sA[tid + st]; sB[tid] += sB[tid + st]; }
    __syncthreads();
  }
  float mu  = sA[0] * (1.0f / DMODEL);
  float var = sB[0] * (1.0f / DMODEL) - mu * mu;
  float rs  = rsqrtf(var + EPS_LN);
  _Float16* orow = o + (size_t)row * DMODEL;
  orow[tid]       = (_Float16)((v0 - mu) * rs * g[tid]);
  orow[tid + 128] = (_Float16)((v1 - mu) * rs * g[tid + 128]);
  orow[tid + 256] = (_Float16)((v2 - mu) * rs * g[tid + 256]);
}

// ---------------------------------------------------------------------------
// Residual + pair-merge + LayerNorm over 768 (writes f16), one block per row.
// ---------------------------------------------------------------------------
__global__ void ln768_res_f16(const float* __restrict__ clusters,
                              const float* __restrict__ res_scale,
                              const float* __restrict__ c2o,
                              const float* __restrict__ g2,
                              _Float16* __restrict__ o) {
  __shared__ float sA[256], sB[256];
  int row = blockIdx.x;
  int bb = row / NPJ, j = row % NPJ;
  int tid = threadIdx.x;
  float v[3];
#pragma unroll
  for (int i = 0; i < 3; ++i) {
    int e = tid + i * 256;
    int sub = e / DMODEL, idx = e - sub * DMODEL;
    size_t src = ((size_t)(bb * NC + 2 * j + sub)) * DMODEL + idx;
    v[i] = clusters[src] * res_scale[idx] + c2o[src];
  }
  sA[tid] = v[0] + v[1] + v[2];
  sB[tid] = v[0] * v[0] + v[1] * v[1] + v[2] * v[2];
  __syncthreads();
  for (int st = 128; st > 0; st >>= 1) {
    if (tid < st) { sA[tid] += sA[tid + st]; sB[tid] += sB[tid + st]; }
    __syncthreads();
  }
  float mu  = sA[0] * (1.0f / DPROJ);
  float var = sB[0] * (1.0f / DPROJ) - mu * mu;
  float rs  = rsqrtf(var + EPS_LN);
  _Float16* orow = o + (size_t)row * DPROJ;
#pragma unroll
  for (int i = 0; i < 3; ++i) {
    int e = tid + i * 256;
    orow[e] = (_Float16)((v[i] - mu) * rs * g2[e]);
  }
}

// ---------------------------------------------------------------------------
// Weight convert + transpose: W f32 [K,N] -> Wt f16 [N,K].
// ---------------------------------------------------------------------------
__global__ void w_to_f16t(const float* __restrict__ W, _Float16* __restrict__ Wt,
                          int K, int N) {
  int t = blockIdx.x * blockDim.x + threadIdx.x;
  if (t >= K * N) return;
  int k = t / N, nn = t - k * N;
  Wt[(size_t)nn * K + k] = (_Float16)W[t];
}

// v_t[b,h,dh,key] from kv[b,key, 384 + h*32 + dh]
__global__ void pack_vt(const _Float16* __restrict__ kvh, _Float16* __restrict__ vt) {
  int t = blockIdx.x * blockDim.x + threadIdx.x;
  if (t >= NB * NH * DH * NTOK) return;
  int key = t % NTOK;
  int rest = t / NTOK;
  int dhi = rest % DH; rest /= DH;
  int hh = rest % NH;  int bb = rest / NH;
  vt[t] = kvh[((size_t)(bb * NTOK + key)) * (2 * DMODEL) + DMODEL + hh * DH + dhi];
}

__global__ void zerof(float* __restrict__ p, int n) {
  int t = blockIdx.x * blockDim.x + threadIdx.x;
  if (t < n) p[t] = 0.f;
}

__global__ void ts_final(const float* __restrict__ ts, float* __restrict__ o) {
  int t = blockIdx.x * blockDim.x + threadIdx.x;
  if (t >= NB * NPJ) return;
  int bb = t / NPJ, j = t - bb * NPJ;
  o[t] = (ts[(size_t)bb * NC + 2 * j] + ts[(size_t)bb * NC + 2 * j + 1]) * (1.0f / NH);
}

// ---------------------------------------------------------------------------
// GEMM: C[M,N] = A_f16[M,K] @ Bt_f16[N,K]^T.
// Block = 128 threads = 4 waves covering 64(M) x 128(N); each wave register-
// blocks a 32x64 tile (2 A frags x 4 B frags -> 8 WMMAs per 12 b128 loads).
// ---------------------------------------------------------------------------
__global__ void gemm_wmma(const _Float16* __restrict__ A,
                          const _Float16* __restrict__ Bt,
                          float* __restrict__ Cf, _Float16* __restrict__ Ch,
                          int K, int lda, int ldb, int ldc,
                          float scale, int store_half) {
  int lane = threadIdx.x & 31;
  int wv   = threadIdx.x >> 5;
  int hi   = lane >> 4;
  int ln15 = lane & 15;
  int mw = blockIdx.x * 64 + (wv & 1) * 32;   // wave's 32-row strip
  int nw = blockIdx.y * 128 + (wv >> 1) * 64; // wave's 64-col strip

  const _Float16* arow0 = A + (size_t)(mw + ln15) * lda + (hi ? 8 : 0);
  const _Float16* arow1 = arow0 + (size_t)16 * lda;
  const _Float16* brow = Bt + (size_t)(nw + ln15) * ldb + (hi ? 8 : 0);
  size_t bstep = (size_t)16 * ldb;

  v8f acc[2][4];
#pragma unroll
  for (int mm = 0; mm < 2; ++mm)
#pragma unroll
    for (int nn = 0; nn < 4; ++nn) acc[mm][nn] = zero8();

  for (int k = 0; k < K; k += 32) {
    v16h a0 = mk16(*(const v8h*)(arow0 + k), *(const v8h*)(arow0 + k + 16));
    v16h a1 = mk16(*(const v8h*)(arow1 + k), *(const v8h*)(arow1 + k + 16));
    v16h b[4];
#pragma unroll
    for (int nn = 0; nn < 4; ++nn) {
      const _Float16* bp = brow + nn * bstep + k;
      b[nn] = mk16(*(const v8h*)bp, *(const v8h*)(bp + 16));
    }
#pragma unroll
    for (int nn = 0; nn < 4; ++nn) {
      acc[0][nn] = wmma_f16(a0, b[nn], acc[0][nn]);
      acc[1][nn] = wmma_f16(a1, b[nn], acc[1][nn]);
    }
  }
#pragma unroll
  for (int mm = 0; mm < 2; ++mm) {
#pragma unroll
    for (int nn = 0; nn < 4; ++nn) {
      int col = nw + nn * 16 + ln15;
#pragma unroll
      for (int r = 0; r < 8; ++r) {
        int rowi = mw + mm * 16 + r + 8 * hi;
        float v = acc[mm][nn][r] * scale;
        if (store_half) Ch[(size_t)rowi * ldc + col] = (_Float16)v;
        else            Cf[(size_t)rowi * ldc + col] = v;
      }
    }
  }
}

// ---------------------------------------------------------------------------
// Pass 1: per-key softmax stats over the cluster axis.
// One wave owns 64 keys of one (b,h): 4 resident K^T fragments; each streamed
// Q tile feeds 4 WMMAs (64 FLOP per L2 byte). Lane = key column in D layout,
// so the online max/sum over clusters is lane-local + one shfl_xor(16).
// ---------------------------------------------------------------------------
__global__ void attn_stats(const _Float16* __restrict__ qh,
                           const _Float16* __restrict__ kvh,
                           float* __restrict__ mbuf, float* __restrict__ zbuf) {
  int lane = threadIdx.x & 31;
  int hi = lane >> 4, ln15 = lane & 15;
  int key0 = blockIdx.x * 64;
  int hh = blockIdx.y, bb = blockIdx.z;

  v16h bk[4];
#pragma unroll
  for (int j = 0; j < 4; ++j) {
    const _Float16* kp = kvh
        + ((size_t)(bb * NTOK + key0 + j * 16 + ln15)) * (2 * DMODEL)
        + hh * DH + (hi ? 8 : 0);
    bk[j] = mk16(*(const v8h*)kp, *(const v8h*)(kp + 16));
  }
  const _Float16* qbase = qh + ((size_t)bb * NC + ln15) * DMODEL + hh * DH + (hi ? 8 : 0);

  float m[4], z[4];
#pragma unroll
  for (int j = 0; j < 4; ++j) { m[j] = -3.0e38f; z[j] = 0.f; }

  for (int ct = 0; ct < NC / 16; ++ct) {
    const _Float16* qp = qbase + (size_t)ct * 16 * DMODEL;
    v16h a = mk16(*(const v8h*)qp, *(const v8h*)(qp + 16));
#pragma unroll
    for (int j = 0; j < 4; ++j) {
      v8f s = wmma_f16(a, bk[j], zero8());
      float tm = s[0];
#pragma unroll
      for (int r = 1; r < 8; ++r) tm = fmaxf(tm, s[r]);
      float nm = fmaxf(m[j], tm);
      float e = 0.f;
#pragma unroll
      for (int r = 0; r < 8; ++r) e += __expf(s[r] - nm);
      z[j] = z[j] * __expf(m[j] - nm) + e;
      m[j] = nm;
    }
  }
#pragma unroll
  for (int j = 0; j < 4; ++j) {
    float m2 = __shfl_xor(m[j], 16);
    float z2 = __shfl_xor(z[j], 16);
    float M = fmaxf(m[j], m2);
    float Z = z[j] * __expf(m[j] - M) + z2 * __expf(m2 - M);
    if (lane < 16) {
      size_t o = ((size_t)bb * NH + hh) * NTOK + key0 + j * 16 + lane;
      mbuf[o] = M;
      zbuf[o] = Z;
    }
  }
}

// ---------------------------------------------------------------------------
// Pass 2: c2 = (softmax P / rowsum) @ V and token sizes.
// One wave owns 32 clusters of one (b,h) (2 resident Q fragments); streams
// key tiles of 32: 4 S-WMMAs -> exp/normalize -> LDS transpose into the f16
// A layout -> 4 PV-WMMAs vs pre-transposed V^T. K/V fragments and per-key
// m,Z are shared by both cluster fragments.
// ---------------------------------------------------------------------------
__global__ void attn_apply(const _Float16* __restrict__ qh,
                           const _Float16* __restrict__ kvh,
                           const _Float16* __restrict__ vt,
                           const float* __restrict__ mbuf,
                           const float* __restrict__ zbuf,
                           _Float16* __restrict__ c2h,
                           float* __restrict__ tsbuf) {
  __shared__ float pshm[2][16][33];
  __shared__ float tshm[2][16];
  int lane = threadIdx.x & 31;
  int hi = lane >> 4, ln15 = lane & 15;
  int c0 = blockIdx.x * 32;
  int hh = blockIdx.y, bb = blockIdx.z;
  size_t hz = ((size_t)bb * NH + hh) * NTOK;

  v16h aq[2];
#pragma unroll
  for (int ci = 0; ci < 2; ++ci) {
    const _Float16* qp = qh + ((size_t)(bb * NC + c0 + ci * 16 + ln15)) * DMODEL
                            + hh * DH + (hi ? 8 : 0);
    aq[ci] = mk16(*(const v8h*)qp, *(const v8h*)(qp + 16));
  }

  v8f acc[2][2];
#pragma unroll
  for (int ci = 0; ci < 2; ++ci) { acc[ci][0] = zero8(); acc[ci][1] = zero8(); }
  float tsacc[2][8];
#pragma unroll
  for (int ci = 0; ci < 2; ++ci)
#pragma unroll
    for (int r = 0; r < 8; ++r) tsacc[ci][r] = 0.f;

  for (int kt = 0; kt < NTOK / 32; ++kt) {
    int key0 = kt * 32;
#pragma unroll
    for (int sub = 0; sub < 2; ++sub) {
      int key = key0 + sub * 16 + ln15;
      const _Float16* kp = kvh + ((size_t)(bb * NTOK + key)) * (2 * DMODEL)
                              + hh * DH + (hi ? 8 : 0);
      v16h bk = mk16(*(const v8h*)kp, *(const v8h*)(kp + 16));
      float mk = mbuf[hz + key];
      float rz = 1.0f / zbuf[hz + key];
#pragma unroll
      for (int ci = 0; ci < 2; ++ci) {
        v8f s = wmma_f16(aq[ci], bk, zero8());
#pragma unroll
        for (int r = 0; r < 8; ++r) {
          float p = __expf(s[r] - mk) * rz;
          tsacc[ci][r] += p;
          pshm[ci][r + 8 * hi][ln15 + 16 * sub] = p;   // [cluster][key]
        }
      }
    }
    // read back in WMMA A layout (M=cluster across lanes, K=keys in regs)
    v16h pa[2];
    int k0 = hi ? 8 : 0;
#pragma unroll
    for (int ci = 0; ci < 2; ++ci)
#pragma unroll
      for (int i = 0; i < 8; ++i) {
        pa[ci][i]     = (_Float16)pshm[ci][ln15][k0 + i];
        pa[ci][i + 8] = (_Float16)pshm[ci][ln15][k0 + 16 + i];
      }
#pragma unroll
    for (int nh = 0; nh < 2; ++nh) {
      const _Float16* vp = vt
          + ((size_t)((bb * NH + hh) * DH + nh * 16 + ln15)) * NTOK
          + key0 + (hi ? 8 : 0);
      v16h bv = mk16(*(const v8h*)vp, *(const v8h*)(vp + 16));
      acc[0][nh] = wmma_f16(pa[0], bv, acc[0][nh]);
      acc[1][nh] = wmma_f16(pa[1], bv, acc[1][nh]);
    }
  }

  // token-size reduction across the 16 key residues (butterfly within halves)
#pragma unroll
  for (int ci = 0; ci < 2; ++ci)
#pragma unroll
    for (int r = 0; r < 8; ++r) {
#pragma unroll
      for (int msk = 1; msk < 16; msk <<= 1)
        tsacc[ci][r] += __shfl_xor(tsacc[ci][r], msk);
    }
  if (ln15 == 0) {
#pragma unroll
    for (int ci = 0; ci < 2; ++ci)
#pragma unroll
      for (int r = 0; r < 8; ++r) {
        tshm[ci][r + 8 * hi] = tsacc[ci][r];
        atomicAdd(&tsbuf[(size_t)bb * NC + c0 + ci * 16 + r + 8 * hi], tsacc[ci][r]);
      }
  }
  // same wave: DS ops are program-ordered; normalize and store c2
#pragma unroll
  for (int ci = 0; ci < 2; ++ci)
#pragma unroll
    for (int r = 0; r < 8; ++r) {
      float inv = 1.0f / (tshm[ci][r + 8 * hi] + EPS_TS);
      size_t o = ((size_t)(bb * NC + c0 + ci * 16 + r + 8 * hi)) * DMODEL
               + hh * DH + ln15;
      c2h[o]      = (_Float16)(acc[ci][0][r] * inv);
      c2h[o + 16] = (_Float16)(acc[ci][1][r] * inv);
    }
}

// ---------------------------------------------------------------------------
extern "C" void kernel_launch(void* const* d_in, const int* in_sizes, int n_in,
                              void* d_out, int out_size, void* d_ws, size_t ws_size,
                              hipStream_t stream) {
  const float* x        = (const float*)d_in[0];
  const float* clusters = (const float*)d_in[1];
  const float* g1       = (const float*)d_in[2];
  const float* Wq       = (const float*)d_in[3];
  const float* Wkv      = (const float*)d_in[4];
  const float* Wo       = (const float*)d_in[5];
  const float* res_scl  = (const float*)d_in[6];
  const float* g2       = (const float*)d_in[7];
  const float* Wproj    = (const float*)d_in[8];
  float* out = (float*)d_out;

  char* ws = (char*)d_ws;
  size_t off = 0;
  auto take = [&](size_t bytes) -> char* {
    char* p = ws + off;
    off += (bytes + 255) & ~(size_t)255;
    return p;
  };
  _Float16* xn     = (_Float16*)take((size_t)NB * NTOK * DMODEL * 2);
  _Float16* cn     = (_Float16*)take((size_t)NB * NC * DMODEL * 2);
  _Float16* wqt    = (_Float16*)take((size_t)DMODEL * DMODEL * 2);
  _Float16* wkvt   = (_Float16*)take((size_t)DMODEL * 2 * DMODEL * 2);
  _Float16* wot    = (_Float16*)take((size_t)DMODEL * DMODEL * 2);
  _Float16* wpjt   = (_Float16*)take((size_t)DPROJ * DPROJ * 2);
  _Float16* q16    = (_Float16*)take((size_t)NB * NC * DMODEL * 2);
  _Float16* kv16   = (_Float16*)take((size_t)NB * NTOK * 2 * DMODEL * 2);
  _Float16* vth    = (_Float16*)take((size_t)NB * NH * DH * NTOK * 2);
  float*    mbuf   = (float*)take((size_t)NB * NH * NTOK * 4);
  float*    zbuf   = (float*)take((size_t)NB * NH * NTOK * 4);
  _Float16* c2h    = (_Float16*)take((size_t)NB * NC * DMODEL * 2);
  float*    tsbuf  = (float*)take((size_t)NB * NC * 4);
  float*    c2o    = (float*)take((size_t)NB * NC * DMODEL * 4);
  _Float16* yn     = (_Float16*)take((size_t)NB * NPJ * DPROJ * 2);

  // 1) LayerNorms -> f16
  ln384_f16<<<NB * NTOK, 128, 0, stream>>>(x, g1, xn);
  ln384_f16<<<NB * NC, 128, 0, stream>>>(clusters, g1, cn);

  // 2) Weight convert + transpose to [N,K] f16
  w_to_f16t<<<(DMODEL * DMODEL + 255) / 256, 256, 0, stream>>>(Wq, wqt, DMODEL, DMODEL);
  w_to_f16t<<<(DMODEL * 2 * DMODEL + 255) / 256, 256, 0, stream>>>(Wkv, wkvt, DMODEL, 2 * DMODEL);
  w_to_f16t<<<(DMODEL * DMODEL + 255) / 256, 256, 0, stream>>>(Wo, wot, DMODEL, DMODEL);
  w_to_f16t<<<(DPROJ * DPROJ + 255) / 256, 256, 0, stream>>>(Wproj, wpjt, DPROJ, DPROJ);

  // 3) Projections (WMMA). Q gets the softmax scale folded in.
  gemm_wmma<<<dim3((NB * NC) / 64, DMODEL / 128), 128, 0, stream>>>(
      cn, wqt, (float*)nullptr, q16, DMODEL, DMODEL, DMODEL, DMODEL, QSCALE, 1);
  gemm_wmma<<<dim3((NB * NTOK) / 64, (2 * DMODEL) / 128), 128, 0, stream>>>(
      xn, wkvt, (float*)nullptr, kv16, DMODEL, DMODEL, DMODEL, 2 * DMODEL, 1.0f, 1);
  pack_vt<<<(NB * NH * DH * NTOK + 255) / 256, 256, 0, stream>>>(kv16, vth);

  // 4) Attention (two passes, streaming; never materializes [b,h,nc,n])
  zerof<<<(NB * NC + 255) / 256, 256, 0, stream>>>(tsbuf, NB * NC);
  attn_stats<<<dim3(NTOK / 64, NH, NB), 32, 0, stream>>>(q16, kv16, mbuf, zbuf);
  attn_apply<<<dim3(NC / 32, NH, NB), 32, 0, stream>>>(q16, kv16, vth,
                                                       mbuf, zbuf, c2h, tsbuf);

  // 5) Output projection Wo, residual + merged LN, Wproj -> d_out
  gemm_wmma<<<dim3((NB * NC) / 64, DMODEL / 128), 128, 0, stream>>>(
      c2h, wot, c2o, (_Float16*)nullptr, DMODEL, DMODEL, DMODEL, DMODEL, 1.0f, 0);
  ln768_res_f16<<<NB * NPJ, 256, 0, stream>>>(clusters, res_scl, c2o, g2, yn);
  gemm_wmma<<<dim3((NB * NPJ) / 64, DPROJ / 128), 128, 0, stream>>>(
      yn, wpjt, out, (_Float16*)nullptr, DPROJ, DPROJ, DPROJ, DPROJ, 1.0f, 0);

  // 6) token_sizes tail of d_out
  ts_final<<<(NB * NPJ + 255) / 256, 256, 0, stream>>>(tsbuf, out + (size_t)NB * NPJ * DPROJ);
}
